// RQVAE_17772574671155
// MI455X (gfx1250) — matrix operand order repaired and checked by reading, not compiled, not run on hardware
//
#include <hip/hip_runtime.h>

typedef __bf16 bf16;
typedef __attribute__((ext_vector_type(16))) __bf16 v16bf;
typedef __attribute__((ext_vector_type(8)))  float  v8f;
typedef __attribute__((ext_vector_type(4)))  int    v4i;

#define BM 128
#define BN 128
#define BK 32
#define LDSS 40   // 32 + 8 bf16 pad -> 80B row stride, 16B aligned, bank-skewed

// ---- CDNA5 async global->LDS copy (ASYNCcnt path), with safe fallback -----
#if __has_builtin(__builtin_amdgcn_global_load_async_to_lds_b128)
#define HAS_ASYNC_LDS 1
typedef __attribute__((address_space(3))) v4i* as3_v4i;
// param1 (probed from clang diagnostic): generic v4i* global source
// param2: LDS destination, AS(3) v4i* via ptrtoint/inttoptr (low 32 bits of a
// generic LDS pointer are the LDS byte offset per the flat-aperture rules)
#define ASYNC_CP(dst, src) \
    __builtin_amdgcn_global_load_async_to_lds_b128( \
        (v4i*)(src), (as3_v4i)(size_t)(dst), 0, 0)
#if __has_builtin(__builtin_amdgcn_s_wait_asynccnt)
#define ASYNC_WAIT() __builtin_amdgcn_s_wait_asynccnt(0)
#else
#define ASYNC_WAIT() asm volatile("s_wait_asynccnt 0" ::: "memory")
#endif
#else
#define ASYNC_CP(dst, src) (*(uint4*)(dst) = *(const uint4*)(src))
#define ASYNC_WAIT()
#endif

// ---------------------------------------------------------------------------
// Tiled bf16 WMMA GEMM: out = epilogue(A[M,K] @ WT[N,K]^T)
// MODE 0: bias + ReLU, bf16 out.  MODE 1: bias, f32 out.
// MODE 2: VQ distance: rnorm[row] + cnorm[col] - 2*acc, f32 out.
// Double-buffered LDS software pipeline, one barrier per K-step.
// ---------------------------------------------------------------------------
template<int MODE>
__global__ __launch_bounds__(256) void gemm_wmma(
    const bf16* __restrict__ A, const bf16* __restrict__ WT,
    const float* __restrict__ bias, void* __restrict__ outp,
    int M, int N, int K, int ldo,
    const float* __restrict__ rnorm, const float* __restrict__ cnorm)
{
    __shared__ bf16 As[2][BM * LDSS];
    __shared__ bf16 Bs[2][BN * LDSS];

    const int tid   = threadIdx.x;
    const int lane  = tid & 31;
    const int wave  = tid >> 5;        // 0..7
    const int waveM = wave >> 1;       // 0..3  -> 32-row strip
    const int waveN = wave & 1;        // 0..1  -> 64-col strip
    const int l15   = lane & 15;
    const int hsel  = lane >> 4;       // 0 | 1
    const int bm = blockIdx.y * BM;
    const int bn = blockIdx.x * BN;

    // static staging assignment: this thread owns rows ar0, ar1, cols ac0..ac0+7
    const int ar0 = tid >> 2;          // 0..63
    const int ar1 = ar0 + 64;          // 64..127
    const int ac0 = (tid & 3) * 8;

    const bf16* Abase = A  + (size_t)bm * K + ac0;
    const bf16* Wbase = WT + (size_t)bn * K + ac0;

    v8f acc[2][4];
    { v8f z = {}; for (int i = 0; i < 2; ++i) for (int j = 0; j < 4; ++j) acc[i][j] = z; }

    // prologue: stage K-tile 0 into buffer 0
    ASYNC_CP(&As[0][ar0 * LDSS + ac0], Abase + (size_t)ar0 * K);
    ASYNC_CP(&As[0][ar1 * LDSS + ac0], Abase + (size_t)ar1 * K);
    ASYNC_CP(&Bs[0][ar0 * LDSS + ac0], Wbase + (size_t)ar0 * K);
    ASYNC_CP(&Bs[0][ar1 * LDSS + ac0], Wbase + (size_t)ar1 * K);
    ASYNC_WAIT();
    __syncthreads();

    int buf = 0;
    for (int k0 = 0; k0 < K; k0 += BK) {
        const int nk = k0 + BK;
        if (nk < K) {   // stream next K-tile into the other buffer
            ASYNC_CP(&As[buf ^ 1][ar0 * LDSS + ac0], Abase + (size_t)ar0 * K + nk);
            ASYNC_CP(&As[buf ^ 1][ar1 * LDSS + ac0], Abase + (size_t)ar1 * K + nk);
            ASYNC_CP(&Bs[buf ^ 1][ar0 * LDSS + ac0], Wbase + (size_t)ar0 * K + nk);
            ASYNC_CP(&Bs[buf ^ 1][ar1 * LDSS + ac0], Wbase + (size_t)ar1 * K + nk);
        }

        const bf16* Asb = As[buf];
        const bf16* Bsb = Bs[buf];

        // A fragment: lanes 0-15 K 0..7 & 16..23 ; lanes 16-31 K 8..15 & 24..31
        v16bf afrag[2];
        for (int tm = 0; tm < 2; ++tm) {
            int r  = waveM * 32 + tm * 16 + l15;
            int kh = hsel * 8;
            union { v16bf v; uint4 q[2]; } u;
            u.q[0] = *(const uint4*)(&Asb[r * LDSS + kh]);
            u.q[1] = *(const uint4*)(&Asb[r * LDSS + kh + 16]);
            afrag[tm] = u.v;
        }
        // B fragment: lane col = l15; lanes 0-15 K 0..7 & 16..23 ; lanes 16-31 K 8..15 & 24..31
        v16bf bfrag[4];
        for (int tn = 0; tn < 4; ++tn) {
            int r  = waveN * 64 + tn * 16 + l15;
            int kh = hsel * 8;
            union { v16bf v; uint4 q[2]; } u;
            u.q[0] = *(const uint4*)(&Bsb[r * LDSS + kh]);
            u.q[1] = *(const uint4*)(&Bsb[r * LDSS + kh + 16]);
            bfrag[tn] = u.v;
        }
        for (int tm = 0; tm < 2; ++tm)
            for (int tn = 0; tn < 4; ++tn)
                acc[tm][tn] = __builtin_amdgcn_wmma_f32_16x16x32_bf16(
                    false, afrag[tm], false, bfrag[tn],
                    (short)0, acc[tm][tn], false, false);

        if (nk < K) {
            ASYNC_WAIT();
            __syncthreads();   // implies ds reads of buf are done in all waves
            buf ^= 1;
        }
    }

    // C/D layout: lane L, vgpr v -> M = v + 8*(L>=16), N = L&15
    for (int tm = 0; tm < 2; ++tm) {
        for (int tn = 0; tn < 4; ++tn) {
            int gc  = bn + waveN * 64 + tn * 16 + l15;
            int gr0 = bm + waveM * 32 + tm * 16 + hsel * 8;
            for (int v = 0; v < 8; ++v) {
                int gr = gr0 + v;
                float val = acc[tm][tn][v];
                if (MODE == 0) {
                    val += bias[gc];
                    val = val > 0.f ? val : 0.f;
                    ((bf16*)outp)[(size_t)gr * ldo + gc] = (bf16)val;
                } else if (MODE == 1) {
                    val += bias[gc];
                    ((float*)outp)[(size_t)gr * ldo + gc] = val;
                } else {
                    val = rnorm[gr] + cnorm[gc] - 2.f * val;
                    ((float*)outp)[(size_t)gr * ldo + gc] = val;
                }
            }
        }
    }
}

// ---------------------------------------------------------------------------
__global__ void zero_f32(float* p, size_t n) {
    size_t i = (size_t)blockIdx.x * 256 + threadIdx.x;
    if (i < n) p[i] = 0.f;
}

__global__ void convert_f32_bf16(const float* __restrict__ in,
                                 bf16* __restrict__ out, size_t n) {
    size_t i = (size_t)blockIdx.x * 256 + threadIdx.x;
    if (i < n) out[i] = (bf16)in[i];
}

// W[K,N] f32 -> WT[N,K] bf16
__global__ void transpose_convert(const float* __restrict__ W,
                                  bf16* __restrict__ WT, int K, int N) {
    size_t idx = (size_t)blockIdx.x * 256 + threadIdx.x;
    if (idx >= (size_t)N * K) return;
    int n = (int)(idx / K), k = (int)(idx % K);
    WT[idx] = (bf16)W[(size_t)k * N + n];
}

// one 128-wide row: f32 -> bf16 copy + row L2 norm (K fixed 128)
__global__ __launch_bounds__(128) void rownorm_convert(
    const float* __restrict__ in, bf16* __restrict__ out,
    float* __restrict__ norm) {
    int b = blockIdx.x, k = threadIdx.x;
    float v = in[(size_t)b * 128 + k];
    out[(size_t)b * 128 + k] = (bf16)v;
    float s = v * v;
    for (int o = 16; o > 0; o >>= 1) s += __shfl_down(s, o, 32);
    __shared__ float red[4];
    if ((k & 31) == 0) red[k >> 5] = s;
    __syncthreads();
    if (k == 0) norm[b] = red[0] + red[1] + red[2] + red[3];
}

// one wave per row: argmin over 1024 cols (first-min), write one-hot + index
__global__ __launch_bounds__(256) void argmin_onehot(
    const float* __restrict__ logits,  // + s*1024, row stride 4096
    float* __restrict__ onehot,        // + s*1024, row stride 4096
    float* __restrict__ idx_f,         // + s, row stride 4
    int* __restrict__ idx_buf) {
    int wv = threadIdx.x >> 5, lane = threadIdx.x & 31;
    int b = blockIdx.x * 8 + wv;
    const float* row = logits + (size_t)b * 4096;
    float best = 3.402823466e38f; int bi = 0;
    for (int j = lane; j < 1024; j += 32) {
        float v = row[j];
        if (v < best) { best = v; bi = j; }
    }
    for (int o = 16; o > 0; o >>= 1) {
        float ov = __shfl_down(best, o, 32);
        int   oi = __shfl_down(bi,   o, 32);
        if (ov < best || (ov == best && oi < bi)) { best = ov; bi = oi; }
    }
    bi = __shfl(bi, 0, 32);
    float* oh = onehot + (size_t)b * 4096;
    for (int j = lane; j < 1024; j += 32) oh[j] = (j == bi) ? 1.f : 0.f;
    if (lane == 0) { idx_f[(size_t)b * 4] = (float)bi; idx_buf[b] = bi; }
}

// gather E[idx], accumulate x_q, new residual (f32 + bf16), next rnorm, SSE
__global__ __launch_bounds__(128) void vq_update(
    const float* __restrict__ E, const int* __restrict__ idx_buf,
    float* __restrict__ residual, bf16* __restrict__ resbf,
    float* __restrict__ rnorm, float* __restrict__ x_q,
    float* __restrict__ loss_acc) {
    int b = blockIdx.x, k = threadIdx.x;
    int idx = idx_buf[b];
    float e = E[(size_t)idx * 128 + k];
    float r = residual[(size_t)b * 128 + k];
    float d = e - r;
    x_q[(size_t)b * 128 + k] += e;
    float nr = r - e;
    residual[(size_t)b * 128 + k] = nr;
    resbf[(size_t)b * 128 + k] = (bf16)nr;
    float s1 = d * d, s2 = nr * nr;
    for (int o = 16; o > 0; o >>= 1) {
        s1 += __shfl_down(s1, o, 32);
        s2 += __shfl_down(s2, o, 32);
    }
    __shared__ float r1[4], r2[4];
    if ((k & 31) == 0) { r1[k >> 5] = s1; r2[k >> 5] = s2; }
    __syncthreads();
    if (k == 0) {
        atomicAdd(loss_acc, r1[0] + r1[1] + r1[2] + r1[3]);
        rnorm[b] = r2[0] + r2[1] + r2[2] + r2[3];
    }
}

__global__ void finalize_loss(const float* __restrict__ acc,
                              float* __restrict__ out) {
    if (threadIdx.x == 0)
        out[0] = acc[0] * 1.25f / (4.0f * 32768.0f * 128.0f);
}

// ---------------------------------------------------------------------------
extern "C" void kernel_launch(void* const* d_in, const int* in_sizes, int n_in,
                              void* d_out, int out_size, void* d_ws, size_t ws_size,
                              hipStream_t stream)
{
    const int B = 32768;
    static const int ENC[6] = {768, 2048, 1024, 512, 256, 128};
    static const int DEC[6] = {128, 256, 512, 1024, 2048, 768};

    const float* x = (const float*)d_in[0];
    const float *enc_w[5], *enc_b[5], *dec_w[5], *dec_b[5];
    for (int i = 0; i < 5; ++i) {
        enc_w[i] = (const float*)d_in[1 + i];
        enc_b[i] = (const float*)d_in[6 + i];
        dec_w[i] = (const float*)d_in[11 + i];
        dec_b[i] = (const float*)d_in[16 + i];
    }
    const float* codebooks = (const float*)d_in[21];

    char* ws = (char*)d_ws;
    size_t off = 0;
    auto alloc = [&](size_t bytes) -> char* {
        char* p = ws + off;
        off = (off + bytes + 255) & ~(size_t)255;
        return p;
    };
    bf16*  actA     = (bf16*) alloc((size_t)B * 2048 * 2);
    bf16*  actB     = (bf16*) alloc((size_t)B * 2048 * 2);
    float* residual = (float*)alloc((size_t)B * 128 * 4);
    float* x_q      = (float*)alloc((size_t)B * 128 * 4);
    bf16*  resbf    = (bf16*) alloc((size_t)B * 128 * 2);
    bf16*  xqbf     = (bf16*) alloc((size_t)B * 128 * 2);
    float* rnorm    = (float*)alloc((size_t)B * 4);
    bf16*  wt[10];
    for (int i = 0; i < 5; ++i) wt[i]     = (bf16*)alloc((size_t)ENC[i] * ENC[i + 1] * 2);
    for (int i = 0; i < 5; ++i) wt[5 + i] = (bf16*)alloc((size_t)DEC[i] * DEC[i + 1] * 2);
    bf16*  cbbf    = (bf16*) alloc((size_t)4 * 1024 * 128 * 2);
    float* enorm   = (float*)alloc((size_t)4096 * 4);
    int*   idxbuf  = (int*)  alloc((size_t)B * 4);
    float* lossacc = (float*)alloc(256);

    float* out_f = (float*)d_out;
    const size_t LOSS_OFF = (size_t)B * 768;
    const size_t IDX_OFF  = LOSS_OFF + 1;
    const size_t OH_OFF   = IDX_OFF + (size_t)B * 4;
    const size_t LG_OFF   = OH_OFF + (size_t)B * 4 * 1024;

    { size_t n = (size_t)B * 128;
      zero_f32<<<(unsigned)((n + 255) / 256), 256, 0, stream>>>(x_q, n); }
    zero_f32<<<1, 32, 0, stream>>>(lossacc, 1);

    for (int i = 0; i < 5; ++i) {
        size_t n = (size_t)ENC[i] * ENC[i + 1];
        transpose_convert<<<(unsigned)((n + 255) / 256), 256, 0, stream>>>(
            enc_w[i], wt[i], ENC[i], ENC[i + 1]);
    }
    for (int i = 0; i < 5; ++i) {
        size_t n = (size_t)DEC[i] * DEC[i + 1];
        transpose_convert<<<(unsigned)((n + 255) / 256), 256, 0, stream>>>(
            dec_w[i], wt[5 + i], DEC[i], DEC[i + 1]);
    }
    { size_t n = (size_t)B * 768;
      convert_f32_bf16<<<(unsigned)((n + 255) / 256), 256, 0, stream>>>(x, actA, n); }

    // ---- encoder ----
    bf16 *cur = actA, *nxt = actB;
    for (int i = 0; i < 4; ++i) {
        int K = ENC[i], N = ENC[i + 1];
        dim3 g(N / 128, B / 128);
        gemm_wmma<0><<<g, 256, 0, stream>>>(cur, wt[i], enc_b[i], nxt,
                                            B, N, K, N, nullptr, nullptr);
        bf16* t = cur; cur = nxt; nxt = t;
    }
    { dim3 g(1, B / 128);
      gemm_wmma<1><<<g, 256, 0, stream>>>(cur, wt[4], enc_b[4], residual,
                                          B, 128, 256, 128, nullptr, nullptr); }

    rownorm_convert<<<B, 128, 0, stream>>>(residual, resbf, rnorm);
    rownorm_convert<<<4096, 128, 0, stream>>>(codebooks, cbbf, enorm);

    // ---- residual VQ stages ----
    for (int s = 0; s < 4; ++s) {
        dim3 g(1024 / 128, B / 128);
        gemm_wmma<2><<<g, 256, 0, stream>>>(
            resbf, cbbf + (size_t)s * 1024 * 128, nullptr,
            out_f + LG_OFF + (size_t)s * 1024, B, 1024, 128, 4096,
            rnorm, enorm + s * 1024);
        argmin_onehot<<<B / 8, 256, 0, stream>>>(
            out_f + LG_OFF + (size_t)s * 1024,
            out_f + OH_OFF + (size_t)s * 1024,
            out_f + IDX_OFF + s, idxbuf);
        vq_update<<<B, 128, 0, stream>>>(
            codebooks + (size_t)s * 1024 * 128, idxbuf,
            residual, resbf, rnorm, x_q, lossacc);
    }

    { size_t n = (size_t)B * 128;
      convert_f32_bf16<<<(unsigned)((n + 255) / 256), 256, 0, stream>>>(x_q, xqbf, n); }

    // ---- decoder ----
    { dim3 g(256 / 128, B / 128);
      gemm_wmma<0><<<g, 256, 0, stream>>>(xqbf, wt[5], dec_b[0], actB,
                                          B, 256, 128, 256, nullptr, nullptr); }
    cur = actB; nxt = actA;
    for (int i = 1; i < 4; ++i) {
        int K = DEC[i], N = DEC[i + 1];
        dim3 g(N / 128, B / 128);
        gemm_wmma<0><<<g, 256, 0, stream>>>(cur, wt[5 + i], dec_b[i], nxt,
                                            B, N, K, N, nullptr, nullptr);
        bf16* t = cur; cur = nxt; nxt = t;
    }
    { dim3 g(768 / 128, B / 128);
      gemm_wmma<1><<<g, 256, 0, stream>>>(cur, wt[9], dec_b[4], out_f,
                                          B, 768, 2048, 768, nullptr, nullptr); }

    finalize_loss<<<1, 32, 0, stream>>>(lossacc, out_f + LOSS_OFF);
}
